// GAT_transformer_18133351924445
// MI455X (gfx1250) — matrix operand order repaired and checked by reading, not compiled, not run on hardware
//
#include <hip/hip_runtime.h>
#include <hip/hip_bf16.h>
#include <math.h>

typedef __attribute__((ext_vector_type(16))) _Float16 v16h;
typedef __attribute__((ext_vector_type(8)))  float    v8f;

#define HEADS 8
#define CSTRIDE 96   // padded concat stride (cols 70..95 stay zero -> K=96 guard-free)

// ---------------------------------------------------------------------------
// helpers: monotone float<->uint key for atomicMax-based segment max
// ---------------------------------------------------------------------------
__device__ __forceinline__ unsigned fkey(float f) {
    unsigned u = __float_as_uint(f);
    return (u & 0x80000000u) ? ~u : (u | 0x80000000u);
}
__device__ __forceinline__ float fdecode(unsigned k) {
    return (k & 0x80000000u) ? __uint_as_float(k ^ 0x80000000u)
                             : __uint_as_float(~k);
}

// ---------------------------------------------------------------------------
// zero init
// ---------------------------------------------------------------------------
__global__ void zero_kernel(float* p, long len) {
    long i = (long)blockIdx.x * blockDim.x + threadIdx.x;
    if (i < len) p[i] = 0.0f;
}

// ---------------------------------------------------------------------------
// Pack B (K x Nc, row-major f32, K zero-padded logically) into WMMA B-fragment
// layout as f16: fragment index (kc*tilesN + nt), 32 lanes x 16 halves each.
// Contiguous k order (same as A loads): lanes 0-15 -> K 0..15, lanes 16-31 -> K 16..31 per chunk,
// N = lane&15, VGPR j holds K pair (2j, 2j+1).
// ---------------------------------------------------------------------------
__global__ void pack_b_kernel(const float* __restrict__ B, _Float16* __restrict__ Bp,
                              int K, int ldb, int tilesN, int kChunks) {
    int idx = (int)blockIdx.x * (int)blockDim.x + (int)threadIdx.x;
    int total = kChunks * tilesN * 32;
    if (idx >= total) return;
    int lane = idx & 31;
    int nt   = (idx >> 5) % tilesN;
    int kc   = (idx >> 5) / tilesN;
    int col  = nt * 16 + (lane & 15);
    int kofB = (lane >> 4) << 4;          // 0 or 16
    _Float16* out = Bp + (size_t)idx * 16;
#pragma unroll
    for (int j = 0; j < 8; ++j) {
        int k0 = kc * 32 + kofB + 2 * j;
        out[2 * j]     = (k0     < K) ? (_Float16)B[(size_t)k0 * ldb + col]       : (_Float16)0.f;
        out[2 * j + 1] = (k0 + 1 < K) ? (_Float16)B[(size_t)(k0 + 1) * ldb + col] : (_Float16)0.f;
    }
}

// ---------------------------------------------------------------------------
// WMMA GEMM strip: each wave computes a 16 x (TN*16) output strip.
// K = KC*32 exactly (caller guarantees zero padding); zero guards in hot loop.
// A: row-major f32 (lda), converted on the fly; B: pre-packed f16 fragments.
// ---------------------------------------------------------------------------
template<int TN, int KC>
__global__ __launch_bounds__(256)
void gemm_wmma_strip(const float* __restrict__ A,
                     const _Float16* __restrict__ Bp,
                     float* __restrict__ C,
                     int M, int lda, int ldc) {
    int wave = ((int)blockIdx.x * (int)blockDim.x + (int)threadIdx.x) >> 5;
    if (wave * 16 >= M) return;
    int lane = threadIdx.x & 31;
    int row  = wave * 16 + (lane & 15);
    int koff = (lane >> 4) << 4;          // 0 or 16

    v8f zero = {};
    v8f acc[TN];
#pragma unroll
    for (int t = 0; t < TN; ++t) acc[t] = zero;

#pragma unroll
    for (int kc = 0; kc < KC; ++kc) {
        const float* pa = A + (size_t)row * lda + kc * 32 + koff;
        v16h a;
#pragma unroll
        for (int j = 0; j < 8; ++j) a[j] = (_Float16)pa[j];          // K kc*32+koff .. +7
#pragma unroll
        for (int j = 0; j < 8; ++j) a[8 + j] = (_Float16)pa[8 + j]; // K +8 .. +15
#pragma unroll
        for (int nt = 0; nt < TN; ++nt) {
            v16h b = *(const v16h*)(Bp + ((size_t)(kc * TN + nt) * 32 + lane) * 16);
            acc[nt] = __builtin_amdgcn_wmma_f32_16x16x32_f16(
                false, a, false, b, (short)0, acc[nt], false, false);
        }
    }

    int mbase = wave * 16 + ((lane >> 4) << 3);
    int nbase = lane & 15;
#pragma unroll
    for (int nt = 0; nt < TN; ++nt)
#pragma unroll
        for (int r = 0; r < 8; ++r)
            C[(size_t)(mbase + r) * ldc + nt * 16 + nbase] = acc[nt][r];
}

// ---------------------------------------------------------------------------
// per-(node,head) attention dot precompute
// ---------------------------------------------------------------------------
__global__ void gat_dots(const float* __restrict__ xw,
                         const float* __restrict__ att,
                         float* __restrict__ ai, float* __restrict__ aj,
                         int N, int C) {
    int idx = (int)blockIdx.x * (int)blockDim.x + (int)threadIdx.x;
    if (idx >= N * HEADS) return;
    int n = idx >> 3, h = idx & 7;
    const float* xr = xw + (size_t)n * (HEADS * C) + h * C;
    const float* ar = att + h * (2 * C);
    float si = 0.f, sj = 0.f;
    for (int c = 0; c < C; ++c) {
        si += xr[c] * ar[c];
        sj += xr[c] * ar[C + c];
    }
    ai[idx] = si;
    aj[idx] = sj;
}

// ---------------------------------------------------------------------------
// edge softmax passes: thread per (edge,head); e in [E, E+N) = self loops
// ---------------------------------------------------------------------------
__device__ __forceinline__ bool edge_sd(const int* src, const int* dst, int e,
                                        int E, int& s, int& d) {
    if (e < E) {
        s = src[e]; d = dst[e];
        return s != d;
    }
    s = d = e - E;
    return true;
}

__global__ void gat_pass_max(const int* __restrict__ src, const int* __restrict__ dst,
                             const float* __restrict__ ai, const float* __restrict__ aj,
                             unsigned* __restrict__ amax, int E, int N) {
    int idx = (int)blockIdx.x * (int)blockDim.x + (int)threadIdx.x;
    if (idx >= (E + N) * HEADS) return;
    int e = idx >> 3, h = idx & 7, s, d;
    if (!edge_sd(src, dst, e, E, s, d)) return;
    float a = ai[d * HEADS + h] + aj[s * HEADS + h];
    a = (a >= 0.f) ? a : 0.2f * a;
    atomicMax(amax + d * HEADS + h, fkey(a));
}

__global__ void gat_pass_sum(const int* __restrict__ src, const int* __restrict__ dst,
                             const float* __restrict__ ai, const float* __restrict__ aj,
                             const unsigned* __restrict__ amax,
                             float* __restrict__ denom, int E, int N) {
    int idx = (int)blockIdx.x * (int)blockDim.x + (int)threadIdx.x;
    if (idx >= (E + N) * HEADS) return;
    int e = idx >> 3, h = idx & 7, s, d;
    if (!edge_sd(src, dst, e, E, s, d)) return;
    float a = ai[d * HEADS + h] + aj[s * HEADS + h];
    a = (a >= 0.f) ? a : 0.2f * a;
    float ea = __expf(a - fdecode(amax[d * HEADS + h]));
    atomicAdd(denom + d * HEADS + h, ea);
}

__global__ void gat_pass_scatter(const int* __restrict__ src, const int* __restrict__ dst,
                                 const float* __restrict__ ai, const float* __restrict__ aj,
                                 const unsigned* __restrict__ amax,
                                 const float* __restrict__ denom,
                                 const float* __restrict__ xw,
                                 float* __restrict__ accum,
                                 int E, int N, int C, int ldacc) {
    int idx = (int)blockIdx.x * (int)blockDim.x + (int)threadIdx.x;
    if (idx >= (E + N) * HEADS) return;
    int e = idx >> 3, h = idx & 7, s, d;
    if (!edge_sd(src, dst, e, E, s, d)) return;
    float a = ai[d * HEADS + h] + aj[s * HEADS + h];
    a = (a >= 0.f) ? a : 0.2f * a;
    float ea = __expf(a - fdecode(amax[d * HEADS + h]));
    float w  = ea / denom[d * HEADS + h];
    const float* xs = xw + (size_t)s * (HEADS * C) + h * C;
    float* out = accum + (size_t)d * ldacc + h * C;
    for (int c = 0; c < C; ++c)
        atomicAdd(out + c, xs[c] * w);
}

// ---------------------------------------------------------------------------
// finalize GAT1: h1 = elu(acc + b1) in place (concat cols 0..63, stride 96)
// ---------------------------------------------------------------------------
__global__ void finalize1_kernel(float* __restrict__ concat,
                                 const float* __restrict__ b1, int N) {
    int idx = (int)blockIdx.x * (int)blockDim.x + (int)threadIdx.x;
    if (idx >= N * 64) return;
    int n = idx >> 6, c = idx & 63;
    float v = concat[(size_t)n * CSTRIDE + c] + b1[c];
    concat[(size_t)n * CSTRIDE + c] = (v > 0.f) ? v : (__expf(v) - 1.f);
}

// ---------------------------------------------------------------------------
// per-node transformer encoder: one wave32 per node, lane == sequence pos
// ---------------------------------------------------------------------------
__device__ __forceinline__ void layernorm6(float* v, const float* g, const float* b) {
    float m = 0.f;
#pragma unroll
    for (int d = 0; d < 6; ++d) m += v[d];
    m *= (1.f / 6.f);
    float var = 0.f;
#pragma unroll
    for (int d = 0; d < 6; ++d) { float t = v[d] - m; var += t * t; }
    var *= (1.f / 6.f);
    float inv = rsqrtf(var + 1e-5f);
#pragma unroll
    for (int d = 0; d < 6; ++d) v[d] = (v[d] - m) * inv * g[d] + b[d];
}

__global__ void encoder_kernel(const float* __restrict__ EF,
                               const float* __restrict__ Wq, const float* __restrict__ bq,
                               const float* __restrict__ Wk, const float* __restrict__ bk,
                               const float* __restrict__ Wv, const float* __restrict__ bv,
                               const float* __restrict__ g0, const float* __restrict__ be0,
                               const float* __restrict__ Wf1, const float* __restrict__ bf1,
                               const float* __restrict__ Wf2, const float* __restrict__ bf2,
                               const float* __restrict__ g1, const float* __restrict__ be1,
                               const float* __restrict__ Wrep, const float* __restrict__ brep,
                               float* __restrict__ concat, int N) {
    __shared__ float sK[8][32][6];
    __shared__ float sV[8][32][6];
    int wv   = threadIdx.x >> 5;
    int s    = threadIdx.x & 31;
    int node = (int)blockIdx.x * 8 + wv;
    bool act = node < N;

    float x[10];
#pragma unroll
    for (int i = 0; i < 10; ++i) x[i] = 0.f;
    if (act) {
        const float* p = EF + ((size_t)node * 32 + s) * 10;
#pragma unroll
        for (int i = 0; i < 10; ++i) x[i] = p[i];
    }

    float Q[6], Kx[6], Vx[6];
#pragma unroll
    for (int d = 0; d < 6; ++d) { Q[d] = bq[d]; Kx[d] = bk[d]; Vx[d] = bv[d]; }
#pragma unroll
    for (int i = 0; i < 10; ++i) {
#pragma unroll
        for (int d = 0; d < 6; ++d) {
            Q[d]  += x[i] * Wq[i * 6 + d];
            Kx[d] += x[i] * Wk[i * 6 + d];
            Vx[d] += x[i] * Wv[i * 6 + d];
        }
    }
#pragma unroll
    for (int d = 0; d < 6; ++d) { sK[wv][s][d] = Kx[d]; sV[wv][s][d] = Vx[d]; }
    __syncthreads();

    const float scale = 0.408248290463863f;  // 1/sqrt(DM=6)
    float O6[6];
#pragma unroll
    for (int h = 0; h < 2; ++h) {
        float sc[32];
        float mx = -1e30f;
        for (int t = 0; t < 32; ++t) {
            float v = Q[h * 3 + 0] * sK[wv][t][h * 3 + 0]
                    + Q[h * 3 + 1] * sK[wv][t][h * 3 + 1]
                    + Q[h * 3 + 2] * sK[wv][t][h * 3 + 2];
            v *= scale;
            sc[t] = v;
            mx = fmaxf(mx, v);
        }
        float sum = 0.f;
        for (int t = 0; t < 32; ++t) { float e = __expf(sc[t] - mx); sc[t] = e; sum += e; }
        float inv = 1.f / sum;
#pragma unroll
        for (int d = 0; d < 3; ++d) {
            float a1 = 0.f;
            for (int t = 0; t < 32; ++t) a1 += sc[t] * sV[wv][t][h * 3 + d];
            O6[h * 3 + d] = Q[h * 3 + d] + a1 * inv;
        }
    }
    layernorm6(O6, g0, be0);

    float ff[6];
#pragma unroll
    for (int d = 0; d < 6; ++d) ff[d] = bf2[d];
    for (int j = 0; j < 24; ++j) {
        float t = bf1[j];
#pragma unroll
        for (int d = 0; d < 6; ++d) t += O6[d] * Wf1[d * 24 + j];
        t = fmaxf(t, 0.f);
#pragma unroll
        for (int d = 0; d < 6; ++d) ff[d] += t * Wf2[j * 6 + d];
    }
    float R[6];
#pragma unroll
    for (int d = 0; d < 6; ++d) R[d] = O6[d] + ff[d];
    layernorm6(R, g1, be1);

    float wr = Wrep[s];
#pragma unroll
    for (int d = 0; d < 6; ++d) {
        float v = R[d] * wr;
#pragma unroll
        for (int off = 16; off >= 1; off >>= 1)
            v += __shfl_xor(v, off, 32);
        if (s == 0 && act) concat[(size_t)node * CSTRIDE + 64 + d] = v + brep[0];
    }
}

// ---------------------------------------------------------------------------
// finalize GAT2: mean over heads + b2, then log_softmax -> d_out
// ---------------------------------------------------------------------------
__global__ void finalize2_kernel(const float* __restrict__ acc2,
                                 const float* __restrict__ b2,
                                 float* __restrict__ out, int N) {
    int n = (int)blockIdx.x * (int)blockDim.x + (int)threadIdx.x;
    if (n >= N) return;
    float v[10];
#pragma unroll
    for (int c = 0; c < 10; ++c) {
        float s = 0.f;
#pragma unroll
        for (int h = 0; h < HEADS; ++h) s += acc2[(size_t)n * 80 + h * 10 + c];
        v[c] = s * 0.125f + b2[c];
    }
    float m = v[0];
#pragma unroll
    for (int c = 1; c < 10; ++c) m = fmaxf(m, v[c]);
    float sum = 0.f;
#pragma unroll
    for (int c = 0; c < 10; ++c) sum += __expf(v[c] - m);
    float lse = __logf(sum);
#pragma unroll
    for (int c = 0; c < 10; ++c) out[n * 10 + c] = v[c] - m - lse;
}

// ---------------------------------------------------------------------------
// host launch
// ---------------------------------------------------------------------------
extern "C" void kernel_launch(void* const* d_in, const int* in_sizes, int n_in,
                              void* d_out, int out_size, void* d_ws, size_t ws_size,
                              hipStream_t stream) {
    const float* x     = (const float*)d_in[0];
    const int*   eidx  = (const int*)  d_in[1];
    const float* efeat = (const float*)d_in[2];
    const float* W1    = (const float*)d_in[3];
    const float* att1  = (const float*)d_in[4];
    const float* b1    = (const float*)d_in[5];
    const float* W2    = (const float*)d_in[6];
    const float* att2  = (const float*)d_in[7];
    const float* b2    = (const float*)d_in[8];
    const float* Wq    = (const float*)d_in[9];
    const float* bq    = (const float*)d_in[10];
    const float* Wk    = (const float*)d_in[11];
    const float* bk    = (const float*)d_in[12];
    const float* Wv    = (const float*)d_in[13];
    const float* bv    = (const float*)d_in[14];
    const float* g0    = (const float*)d_in[15];
    const float* be0   = (const float*)d_in[16];
    const float* Wf1   = (const float*)d_in[17];
    const float* bf1   = (const float*)d_in[18];
    const float* Wf2   = (const float*)d_in[19];
    const float* bf2   = (const float*)d_in[20];
    const float* g1    = (const float*)d_in[21];
    const float* be1   = (const float*)d_in[22];
    const float* Wrep  = (const float*)d_in[23];
    const float* brep  = (const float*)d_in[24];

    const int N = in_sizes[0] / 64;       // 100000 (multiple of 16)
    const int E = in_sizes[1] / 2;        // 1600000
    const int* src = eidx;
    const int* dst = eidx + E;

    // workspace layout (floats); zeroed prefix = concat .. acc2
    float* ws = (float*)d_ws;
    float*    concat = ws;                                    // N*96 (padded)
    unsigned* amax1  = (unsigned*)(ws + (size_t)N * 96);      // N*8
    float*    denom1 = ws + (size_t)N * 104;                  // N*8
    unsigned* amax2  = (unsigned*)(ws + (size_t)N * 112);     // N*8
    float*    denom2 = ws + (size_t)N * 120;                  // N*8
    float*    acc2   = ws + (size_t)N * 128;                  // N*80
    float*    xw1    = ws + (size_t)N * 208;                  // N*64
    float*    xw2    = ws + (size_t)N * 272;                  // N*80
    float*    ai1    = ws + (size_t)N * 352;                  // N*8
    float*    aj1    = ws + (size_t)N * 360;                  // N*8
    float*    ai2    = ws + (size_t)N * 368;                  // N*8
    float*    aj2    = ws + (size_t)N * 376;                  // N*8
    _Float16* Bp1    = (_Float16*)(ws + (size_t)N * 384);     // 2*4*512 halves
    _Float16* Bp2    = Bp1 + 2 * 4 * 512;                     // 3*5*512 halves

    const int TB = 256;
    long zlen = (long)N * 208;
    zero_kernel<<<(int)((zlen + TB - 1) / TB), TB, 0, stream>>>(ws, zlen);

    // pack weights into WMMA B fragments (f16)
    pack_b_kernel<<<1, TB, 0, stream>>>(W1, Bp1, 64, 64, 4, 2);   // 256 frags threads
    pack_b_kernel<<<2, TB, 0, stream>>>(W2, Bp2, 70, 80, 5, 3);   // 480 threads

    int mWaves  = N / 16;                 // 6250
    int mBlocks = (mWaves + 7) / 8;

    // --- GAT1 linear: xw1 = x @ W1  (K = 64 = 2 chunks, Nc = 64 = 4 tiles) ---
    gemm_wmma_strip<4, 2><<<mBlocks, TB, 0, stream>>>(x, Bp1, xw1, N, 64, 64);
    gat_dots<<<(N * HEADS + TB - 1) / TB, TB, 0, stream>>>(xw1, att1, ai1, aj1, N, 8);

    int eThreads = (E + N) * HEADS;
    int eBlocks  = (eThreads + TB - 1) / TB;
    gat_pass_max<<<eBlocks, TB, 0, stream>>>(src, dst, ai1, aj1, amax1, E, N);
    gat_pass_sum<<<eBlocks, TB, 0, stream>>>(src, dst, ai1, aj1, amax1, denom1, E, N);
    gat_pass_scatter<<<eBlocks, TB, 0, stream>>>(src, dst, ai1, aj1, amax1, denom1,
                                                 xw1, concat, E, N, 8, CSTRIDE);
    finalize1_kernel<<<(N * 64 + TB - 1) / TB, TB, 0, stream>>>(concat, b1, N);

    // --- encoder -> concat cols 64..69 ---
    encoder_kernel<<<(N + 7) / 8, TB, 0, stream>>>(efeat, Wq, bq, Wk, bk, Wv, bv,
                                                   g0, be0, Wf1, bf1, Wf2, bf2,
                                                   g1, be1, Wrep, brep, concat, N);

    // --- GAT2 linear: xw2 = concat @ W2 (K = 96 = 3 chunks incl. zero pad) ---
    gemm_wmma_strip<5, 3><<<mBlocks, TB, 0, stream>>>(concat, Bp2, xw2, N, CSTRIDE, 80);
    gat_dots<<<(N * HEADS + TB - 1) / TB, TB, 0, stream>>>(xw2, att2, ai2, aj2, N, 10);

    gat_pass_max<<<eBlocks, TB, 0, stream>>>(src, dst, ai2, aj2, amax2, E, N);
    gat_pass_sum<<<eBlocks, TB, 0, stream>>>(src, dst, ai2, aj2, amax2, denom2, E, N);
    gat_pass_scatter<<<eBlocks, TB, 0, stream>>>(src, dst, ai2, aj2, amax2, denom2,
                                                 xw2, acc2, E, N, 10, 80);

    finalize2_kernel<<<(N + TB - 1) / TB, TB, 0, stream>>>(acc2, b2, (float*)d_out, N);
}
